// RNN_Bidireccional_80290118631704
// MI455X (gfx1250) — matrix operand order repaired and checked by reading, not compiled, not run on hardware
//
#include <hip/hip_runtime.h>
#include <hip/hip_bf16.h>
#include <math.h>

// ---------------------------------------------------------------------------
// Problem constants (from reference): B=64, T=1024, V=50000, E=512, H=256, L=2
// ---------------------------------------------------------------------------
#define B_   64
#define T_   1024
#define E_   512
#define H_   256
#define I_   512          // input width of both RNN layers (E == 2H == 512)

typedef __attribute__((ext_vector_type(16))) __bf16 v16bf;
typedef __attribute__((ext_vector_type(8)))  __bf16 v8bf;
typedef __attribute__((ext_vector_type(8)))  float  v8f;

// ---------------------------------------------------------------------------
// WMMA tile loaders (layouts per CDNA5 ISA 7.12.2, 16-bit A 16x32 / B 32x16)
// ---------------------------------------------------------------------------
// A tile: lane holds row m = lane&15; elements 0..7 = K[koff..koff+7],
// elements 8..15 = K[16+koff..16+koff+7], koff = (lane>>4)*8.
__device__ __forceinline__ v16bf load_a_tile(const __bf16* __restrict__ row,
                                             int kbase, int lane) {
  const int koff = (lane >> 4) * 8;
  v8bf lo = *(const v8bf*)(row + kbase + koff);
  v8bf hi = *(const v8bf*)(row + kbase + 16 + koff);
  v16bf a;
#pragma unroll
  for (int j = 0; j < 8; ++j) { a[j] = lo[j]; a[j + 8] = hi[j]; }
  return a;
}

// Same, with ReLU applied elementwise (used for the head projections).
__device__ __forceinline__ v16bf load_a_tile_relu(const __bf16* __restrict__ row,
                                                  int kbase, int lane) {
  const int koff = (lane >> 4) * 8;
  v8bf lo = *(const v8bf*)(row + kbase + koff);
  v8bf hi = *(const v8bf*)(row + kbase + 16 + koff);
  v16bf a;
#pragma unroll
  for (int j = 0; j < 8; ++j) {
    float fl = (float)lo[j], fh = (float)hi[j];
    a[j]     = (__bf16)(fl > 0.f ? fl : 0.f);
    a[j + 8] = (__bf16)(fh > 0.f ? fh : 0.f);
  }
  return a;
}

// B tile for  D = A * W^T :  B[k][n] = W[n][k]  (W row-major [N][K]).
// Lane holds column n = lane&15; elements 0..7 = K[koff..koff+7],
// elements 8..15 = K[16+koff..16+koff+7], koff = (lane>>4)*8 (two 16-byte loads).
__device__ __forceinline__ v16bf load_bT_tile(const __bf16* __restrict__ wrow,
                                              int kbase, int lane) {
  const int koff = (lane >> 4) * 8;
  v8bf lo = *(const v8bf*)(wrow + kbase + koff);
  v8bf hi = *(const v8bf*)(wrow + kbase + 16 + koff);
  v16bf b;
#pragma unroll
  for (int j = 0; j < 8; ++j) { b[j] = lo[j]; b[j + 8] = hi[j]; }
  return b;
}

// ---------------------------------------------------------------------------
// K0: generic f32 -> bf16 cast
// ---------------------------------------------------------------------------
__global__ void cast_f32_bf16(const float* __restrict__ src,
                              __bf16* __restrict__ dst, int n) {
  int i = blockIdx.x * 256 + threadIdx.x;
  if (i < n) dst[i] = (__bf16)src[i];
}

// ---------------------------------------------------------------------------
// K1: embedding gather + cast:  X[b*T+t][e] = bf16(emb[tokens[b,t]][e])
// ---------------------------------------------------------------------------
__global__ void embed_cast(const int* __restrict__ tokens,
                           const float* __restrict__ emb,
                           __bf16* __restrict__ X) {
  size_t i = (size_t)blockIdx.x * 256 + threadIdx.x;   // over B*T*E
  size_t r = i >> 9;                                   // / 512
  int    e = (int)(i & 511);
  int tok = tokens[r];
  X[i] = (__bf16)emb[(size_t)tok * E_ + e];
}

// ---------------------------------------------------------------------------
// K2: concatenated head weights [16][512] (rows 13..15 zero) + fused biases
// ---------------------------------------------------------------------------
__global__ void build_wcat(const float* wi, const float* wf, const float* wc,
                           const float* bii, const float* bih,
                           const float* bfi, const float* bfh,
                           const float* bci, const float* bch,
                           __bf16* __restrict__ wcat, float* __restrict__ bcat) {
  int i = blockIdx.x * 256 + threadIdx.x;
  if (i < 16 * 512) {
    int c = i >> 9, k = i & 511;
    float v = 0.f;
    if (c < 3)       v = wi[c * 512 + k];
    else if (c < 8)  v = wf[(c - 3) * 512 + k];
    else if (c < 13) v = wc[(c - 8) * 512 + k];
    wcat[i] = (__bf16)v;
  }
  if (i < 16) {
    float v = 0.f;
    if (i < 3)       v = bii[i] + bih[i];
    else if (i < 8)  v = bfi[i - 3] + bfh[i - 3];
    else if (i < 13) v = bci[i - 8] + bch[i - 8];
    bcat[i] = v;
  }
}

// ---------------------------------------------------------------------------
// K3: big input projection  XP[dir][t][b][h] = X[b*T+t] . w_ih[dir][h] + biases
// One 16x16 output tile per wave, K=512 -> 16 WMMAs. 8 waves/block share the
// M-tile (A rows hit L0/L2).  grid = (BT/16, 2, dirs=2), block = 256.
// ---------------------------------------------------------------------------
__global__ void __launch_bounds__(256) proj_big(
    const __bf16* __restrict__ X,    // [B*T][512]
    const __bf16* __restrict__ Wih,  // [2][H][512]  (layer slice, bf16)
    const float*  __restrict__ bih,  // [2][H]
    const float*  __restrict__ bhh,  // [2][H]
    float* __restrict__ XP)          // [2][T][B][H]
{
  const int lane  = threadIdx.x & 31;
  const int wave  = threadIdx.x >> 5;
  const int mtile = blockIdx.x;                 // 0..4095
  const int ntile = blockIdx.y * 8 + wave;      // 0..15
  const int dir   = blockIdx.z;

  const int m = lane & 15;                      // A row within tile / B column
  const __bf16* arow = X + ((size_t)mtile * 16 + m) * I_;
  const int ncol = ntile * 16 + m;
  const __bf16* brow = Wih + ((size_t)dir * H_ + ncol) * I_;

  v8f acc = {};
#pragma unroll 4
  for (int k = 0; k < I_; k += 32) {
    v16bf a = load_a_tile(arow, k, lane);
    v16bf b = load_bT_tile(brow, k, lane);
    acc = __builtin_amdgcn_wmma_f32_16x16x32_bf16(false, a, false, b,
                                                  (short)0, acc, false, false);
  }
  const float bias = bih[dir * H_ + ncol] + bhh[dir * H_ + ncol];
  const size_t Rbase = (size_t)mtile * 16;
#pragma unroll
  for (int r = 0; r < 8; ++r) {
    size_t R = Rbase + (lane >> 4) * 8 + r;     // global row = b*T + t
    int b_ = (int)(R >> 10);                    // / T
    int t  = (int)(R & (T_ - 1));
    XP[(((size_t)dir * T_ + t) * B_ + b_) * H_ + ncol] = acc[r] + bias;
  }
}

// ---------------------------------------------------------------------------
// K4: recurrent scan.  h_t = tanh(xp_t + h_{t-1} W_hh^T), per (dir, b-chunk).
// Critical-path design:
//  - Each wave owns one 16x16 output tile (ntile); its 8 B-tiles of W_hh^T
//    are time-invariant -> hoisted into 64 VGPRs ONCE (no weight traffic in
//    the loop at all; no weights in LDS).
//  - h double-buffered in LDS (16 KB static) -> one barrier per step.
//  - xp_t (the f32 C-accumulator) for step t+1 is prefetched before the WMMA
//    chain of step t so the strided global loads hide under the matrix ops.
// grid = (dirs=2, chunks=4), block = 512 (16 waves).
// ---------------------------------------------------------------------------
__global__ void __launch_bounds__(512) scan_big(
    const float*  __restrict__ XP,    // [2][T][B][H]
    const __bf16* __restrict__ Whh,   // [2][H][H] (layer slice, bf16)
    __bf16* __restrict__ Xout)        // [B*T][512]; dir0 -> cols 0..255, dir1 -> 256..511
{
  __shared__ __bf16 Hsh[2][16][H_];             // double-buffered h (16 KB)

  const int dir   = blockIdx.x;
  const int chunk = blockIdx.y;                 // batch rows chunk*16 .. +15
  const int lane  = threadIdx.x & 31;
  const int ntile = threadIdx.x >> 5;           // 0..15

  const int n    = lane & 15;
  const int ncol = ntile * 16 + n;

  // Hoist this wave's W_hh^T B-tiles into registers (invariant over time).
  v16bf Breg[8];
  {
    const __bf16* brow = Whh + ((size_t)dir * H_ + ncol) * H_;
#pragma unroll
    for (int kt = 0; kt < 8; ++kt)
      Breg[kt] = load_bT_tile(brow, kt * 32, lane);
  }

  // h0 = 0
  for (int i = threadIdx.x * 8; i < 16 * H_; i += 512 * 8) {
    v8bf z = {};
    *(v8bf*)(&Hsh[0][0][0] + i) = z;
  }
  __syncthreads();

  const int mh     = lane & 15;                 // A row within chunk
  const int dircol = dir * H_;
  const int mrow   = (lane >> 4) * 8;           // first C/D row of this lane
  int cur = 0;
  int t = dir ? (T_ - 1) : 0;

  const float* xp0 =
      XP + (size_t)dir * T_ * B_ * H_ + (size_t)(chunk * 16) * H_ + ncol;

  // preload xp for the first step
  v8f accn;
  {
    const float* p = xp0 + (size_t)t * B_ * H_;
#pragma unroll
    for (int r = 0; r < 8; ++r) accn[r] = p[(size_t)(mrow + r) * H_];
  }

  for (int step = 0; step < T_; ++step) {
    v8f acc = accn;

    // prefetch next step's xp (independent of LDS state; hides under WMMAs)
    if (step + 1 < T_) {
      const int tn = dir ? (T_ - 2 - step) : (step + 1);
      const float* p = xp0 + (size_t)tn * B_ * H_;
#pragma unroll
      for (int r = 0; r < 8; ++r) accn[r] = p[(size_t)(mrow + r) * H_];
    }

    const __bf16* arow = &Hsh[cur][mh][0];
#pragma unroll
    for (int kt = 0; kt < 8; ++kt) {
      v16bf a = load_a_tile(arow, kt * 32, lane);
      acc = __builtin_amdgcn_wmma_f32_16x16x32_bf16(false, a, false, Breg[kt],
                                                    (short)0, acc, false, false);
    }

    const int nxt = cur ^ 1;
#pragma unroll
    for (int r = 0; r < 8; ++r) {
      const int M = mrow + r;
      __bf16 hb = (__bf16)tanhf(acc[r]);
      Hsh[nxt][M][ncol] = hb;
      const size_t gb = (size_t)(chunk * 16 + M);
      Xout[(gb * T_ + t) * I_ + dircol + ncol] = hb;
    }
    __syncthreads();          // double buffer: one barrier per step suffices
    cur = nxt;
    t = dir ? (t - 1) : (t + 1);
  }
}

// ---------------------------------------------------------------------------
// K5: head projection  XPS[b*T+t][0..15] = relu(X2 row) . Wcat^T + bcat
// ---------------------------------------------------------------------------
__global__ void __launch_bounds__(256) proj_small(
    const __bf16* __restrict__ X2,    // [B*T][512]
    const __bf16* __restrict__ Wcat,  // [16][512]
    const float*  __restrict__ bcat,  // [16]
    float* __restrict__ XPS)          // [B*T][16]
{
  const int lane  = threadIdx.x & 31;
  const int wave  = threadIdx.x >> 5;
  const int mtile = blockIdx.x * 8 + wave;      // 0..4095
  const int m = lane & 15;
  const __bf16* arow = X2 + ((size_t)mtile * 16 + m) * I_;
  const __bf16* brow = Wcat + (size_t)m * I_;

  v8f acc = {};
#pragma unroll 4
  for (int k = 0; k < I_; k += 32) {
    v16bf a = load_a_tile_relu(arow, k, lane);
    v16bf b = load_bT_tile(brow, k, lane);
    acc = __builtin_amdgcn_wmma_f32_16x16x32_bf16(false, a, false, b,
                                                  (short)0, acc, false, false);
  }
  const float bias = bcat[m];
#pragma unroll
  for (int r = 0; r < 8; ++r) {
    size_t R = (size_t)mtile * 16 + (lane >> 4) * 8 + r;
    XPS[R * 16 + m] = acc[r] + bias;
  }
}

// ---------------------------------------------------------------------------
// K6: tiny head scan — 64 independent per-batch chains per head.
// ---------------------------------------------------------------------------
__global__ void __launch_bounds__(64) scan_small(
    const float* __restrict__ XPS,    // [B*T][16]
    const float* __restrict__ whh,    // [C][C]
    float* __restrict__ out,          // [B][T][C]
    int C, int colbase)
{
  const int b = threadIdx.x;
  float W[25];
  for (int i = 0; i < C * C; ++i) W[i] = whh[i];
  float h[5] = {0.f, 0.f, 0.f, 0.f, 0.f};
  const float* xp = XPS + (size_t)b * T_ * 16;
  float* o = out + (size_t)b * T_ * C;
  for (int t = 0; t < T_; ++t) {
    float s[5];
    for (int c = 0; c < C; ++c) {
      float v = xp[t * 16 + colbase + c];
      for (int j = 0; j < C; ++j) v += W[c * C + j] * h[j];
      s[c] = v;
    }
    for (int c = 0; c < C; ++c) {
      h[c] = tanhf(s[c]);
      o[t * C + c] = h[c];
    }
  }
}

// ---------------------------------------------------------------------------
// Host orchestration
// ---------------------------------------------------------------------------
extern "C" void kernel_launch(void* const* d_in, const int* in_sizes, int n_in,
                              void* d_out, int out_size, void* d_ws, size_t ws_size,
                              hipStream_t stream) {
  (void)in_sizes; (void)n_in; (void)out_size; (void)ws_size;

  const int*   tokens  = (const int*)  d_in[0];
  const float* emb     = (const float*)d_in[1];
  const float* rnn_wih = (const float*)d_in[2];   // [2][2][256][512]
  const float* rnn_whh = (const float*)d_in[3];   // [2][2][256][256]
  const float* rnn_bih = (const float*)d_in[4];   // [2][2][256]
  const float* rnn_bhh = (const float*)d_in[5];
  const float* init_wih = (const float*)d_in[6];
  const float* init_whh = (const float*)d_in[7];
  const float* init_bih = (const float*)d_in[8];
  const float* init_bhh = (const float*)d_in[9];
  const float* fin_wih  = (const float*)d_in[10];
  const float* fin_whh  = (const float*)d_in[11];
  const float* fin_bih  = (const float*)d_in[12];
  const float* fin_bhh  = (const float*)d_in[13];
  const float* cap_wih  = (const float*)d_in[14];
  const float* cap_whh  = (const float*)d_in[15];
  const float* cap_bih  = (const float*)d_in[16];
  const float* cap_bhh  = (const float*)d_in[17];
  float* out = (float*)d_out;

  // ---- workspace layout ----
  char* ws = (char*)d_ws;
  const size_t szX   = (size_t)B_ * T_ * I_ * sizeof(__bf16);       // 64 MB
  const size_t szXP  = (size_t)2 * T_ * B_ * H_ * sizeof(float);    // 128 MB
  const size_t szXPS = (size_t)B_ * T_ * 16 * sizeof(float);        // 4 MB
  const size_t szWih = (size_t)2 * 2 * H_ * I_ * sizeof(__bf16);    // 1 MB
  const size_t szWhh = (size_t)2 * 2 * H_ * H_ * sizeof(__bf16);    // 512 KB
  const size_t szWct = (size_t)16 * I_ * sizeof(__bf16);            // 16 KB

  __bf16* Xa    = (__bf16*)(ws);
  __bf16* Xb    = (__bf16*)(ws + szX);
  float*  XP    = (float*) (ws + 2 * szX);
  float*  XPS   = (float*) (ws + 2 * szX + szXP);
  __bf16* WihB  = (__bf16*)(ws + 2 * szX + szXP + szXPS);
  __bf16* WhhB  = (__bf16*)(ws + 2 * szX + szXP + szXPS + szWih);
  __bf16* Wcat  = (__bf16*)(ws + 2 * szX + szXP + szXPS + szWih + szWhh);
  float*  Bcat  = (float*) (ws + 2 * szX + szXP + szXPS + szWih + szWhh + szWct);

  // ---- weight precision conversion (once per launch) ----
  {
    int n1 = 2 * 2 * H_ * I_;   // 524288
    cast_f32_bf16<<<(n1 + 255) / 256, 256, 0, stream>>>(rnn_wih, WihB, n1);
    int n2 = 2 * 2 * H_ * H_;   // 262144
    cast_f32_bf16<<<(n2 + 255) / 256, 256, 0, stream>>>(rnn_whh, WhhB, n2);
    build_wcat<<<(16 * 512 + 255) / 256, 256, 0, stream>>>(
        init_wih, fin_wih, cap_wih,
        init_bih, init_bhh, fin_bih, fin_bhh, cap_bih, cap_bhh, Wcat, Bcat);
  }

  // ---- embedding gather ----
  {
    size_t n = (size_t)B_ * T_ * E_;
    embed_cast<<<(unsigned)(n / 256), 256, 0, stream>>>(tokens, emb, Xa);
  }

  const int MT = (B_ * T_) / 16;                       // 4096 M-tiles

  // ---- layer 0: Xa -> Xb ----
  proj_big<<<dim3(MT, 2, 2), 256, 0, stream>>>(
      Xa, WihB + 0, rnn_bih + 0, rnn_bhh + 0, XP);
  scan_big<<<dim3(2, 4), 512, 0, stream>>>(XP, WhhB + 0, Xb);

  // ---- layer 1: Xb -> Xa ----
  proj_big<<<dim3(MT, 2, 2), 256, 0, stream>>>(
      Xb, WihB + (size_t)2 * H_ * I_, rnn_bih + 2 * H_, rnn_bhh + 2 * H_, XP);
  scan_big<<<dim3(2, 4), 512, 0, stream>>>(
      XP, WhhB + (size_t)2 * H_ * H_, Xa);

  // ---- heads: relu + projection, then tiny scans ----
  proj_small<<<MT / 8, 256, 0, stream>>>(Xa, Wcat, Bcat, XPS);

  const size_t oInit = 0;
  const size_t oFin  = (size_t)B_ * T_ * 3;
  const size_t oCap  = oFin + (size_t)B_ * T_ * 5;
  scan_small<<<1, 64, 0, stream>>>(XPS, init_whh, out + oInit, 3, 0);
  scan_small<<<1, 64, 0, stream>>>(XPS, fin_whh,  out + oFin,  5, 3);
  scan_small<<<1, 64, 0, stream>>>(XPS, cap_whh,  out + oCap,  5, 8);
}
